// KernelOptimizedGraphNeuralNetwork_15272903704635
// MI455X (gfx1250) — matrix operand order repaired and checked by reading, not compiled, not run on hardware
//
#include <hip/hip_runtime.h>
#include <math.h>

#define NN   20000   // nodes
#define NE   320000  // edges
#define DIM  128
#define NH   4
#define HDM  32
#define NL   4
#define NFI  64
#define NCLS 10

typedef __attribute__((ext_vector_type(16))) _Float16 v16h;
typedef __attribute__((ext_vector_type(8)))  float    v8f;

#if defined(__gfx1250__) && __has_builtin(__builtin_amdgcn_global_load_async_to_lds_b128)
#define USE_ASYNC_LDS 1
typedef int v4i __attribute__((vector_size(16)));
typedef __attribute__((address_space(1))) v4i* gv4i_p;   // clang prints AS1 as __device__
typedef __attribute__((address_space(3))) v4i* lv4i_p;   // AS3 = LDS
#else
#define USE_ASYNC_LDS 0
#endif

// branch-free GELU: 0.5*x*(1+tanh(u)) == x*sigmoid(2u); one v_exp_f32 + v_rcp_f32
__device__ __forceinline__ float gelu_f(float x) {
  const float u2 = 1.5957691216057308f * (x + 0.044715f * x * x * x);
  return x / (1.0f + __expf(-u2));
}

__device__ __forceinline__ void atomicMaxF(float* addr, float val) {
  if (val >= 0.0f) atomicMax((int*)addr, __float_as_int(val));
  else             atomicMin((unsigned int*)addr, __float_as_uint(val));
}

// fragment-order maps (ISA 7.12.2, wave32, 16-bit A 16x32):
// forward: (row,k) -> lane = row + 16*((k>>3)&1), j = (k&7) + 8*(k>>4)
// inverse: (lane,j) -> row = lane&15, k = (j&7) + 16*(j>>3) + 8*(lane>>4)
__device__ __forceinline__ int afrag_lane(int row, int k) { return row + (((k >> 3) & 1) << 4); }
__device__ __forceinline__ int afrag_j(int k) { return (k & 7) + ((k >> 4) << 3); }

// ---------------- utility kernels ----------------

__global__ void k_fill(float* __restrict__ p, float v, int n) {
  int i = blockIdx.x * blockDim.x + threadIdx.x;
  if (i < n) p[i] = v;
}

__global__ void k_copy(const float* __restrict__ s, float* __restrict__ d, int n) {
  int i = blockIdx.x * blockDim.x + threadIdx.x;
  if (i < n) d[i] = s[i];
}

// Swizzle f32 row-major W[K,N] into WMMA-B fragment order (f16); each lane's
// v16h B fragment becomes 32 contiguous bytes.
__global__ void k_swizzle_w(const float* __restrict__ W, _Float16* __restrict__ Wf,
                            int K, int N) {
  int idx = blockIdx.x * blockDim.x + threadIdx.x;
  if (idx >= K * N) return;
  const int frag = idx >> 9;
  const int rem  = idx & 511;
  const int lane = rem >> 4;
  const int j    = rem & 15;
  const int ntiles = N >> 4;
  const int kt = frag / ntiles, nt = frag - kt * ntiles;
  const int k = kt * 32 + ((lane >> 4) << 3) + (j & 7) + ((j >> 3) << 4);
  const int c = nt * 16 + (lane & 15);
  Wf[idx] = (_Float16)W[(size_t)k * N + c];
}

// Convert f32 row-major A[M,K] into WMMA-A fragment order (f16), output-coalesced.
// Layout: Af[((mt*nK + kt)*32 + lane)*16 + j]
__global__ void k_cvt_frag(const float* __restrict__ A, _Float16* __restrict__ Af,
                           int K, int n) {
  int idx = blockIdx.x * blockDim.x + threadIdx.x;
  if (idx >= n) return;
  const int nK = K >> 5;
  const int j = idx & 15, lane = (idx >> 4) & 31;
  const int t = idx >> 9;
  const int kt = t % nK, mt = t / nK;
  const int row = mt * 16 + (lane & 15);
  const int k = kt * 32 + (j & 7) + ((j >> 3) << 4) + ((lane >> 4) << 3);
  Af[idx] = (_Float16)A[(size_t)row * K + k];
}

// ---------------- WMMA GEMM: C[M,128] = Afrag[M,K] @ W[K,128] + bias ----------------
// block=(32,8); MPB M-tiles per block (template => no guards, EXEC all-1s everywhere).
// A: f16 fragment-order global; Wf: pre-swizzled f16; C: f32 (+ optional fragment-order
// f16 copy Cf for the next GEMM). A tiles staged via async copy to LDS when available.
template<int MPB>
__global__ void k_gemm_wmma(const _Float16* __restrict__ Afr, const _Float16* __restrict__ Wf,
                            const float* __restrict__ bias, float* __restrict__ C,
                            _Float16* __restrict__ Cf, int K, int mtBase) {
  const int lane = threadIdx.x, wave = threadIdx.y;
  const int nK = K >> 5;
  const int mt0 = mtBase + blockIdx.x * MPB;
  v8f acc[MPB];
#pragma unroll
  for (int mt = 0; mt < MPB; ++mt) acc[mt] = (v8f){0,0,0,0,0,0,0,0};

#if USE_ASYNC_LDS
  __shared__ _Float16 Asm[2][MPB * 512];
  const int tid = wave * 32 + lane;
  auto issue = [&](int buf, int kt) {
    if (tid < MPB * 64) {   // wave-granular split: no intra-wave divergence
      const _Float16* src = Afr + ((size_t)(mt0 + (tid >> 6)) * nK + kt) * 512 + (tid & 63) * 8;
      _Float16* dst = &Asm[buf][(tid >> 6) * 512 + (tid & 63) * 8];
      __builtin_amdgcn_global_load_async_to_lds_b128(
          (gv4i_p)(unsigned long long)(size_t)src,
          (lv4i_p)(unsigned int)(size_t)dst, 0, 0);
    }
  };
  issue(0, 0);
  for (int kt = 0; kt < nK; ++kt) {
    if (kt + 1 < nK) {
      issue((kt + 1) & 1, kt + 1);
      asm volatile("s_wait_asynccnt 0x1" ::: "memory");
    } else {
      asm volatile("s_wait_asynccnt 0x0" ::: "memory");
    }
    __syncthreads();
    if (kt + 1 < nK)
      __builtin_prefetch(&Wf[((size_t)((kt + 1) * 8 + wave) * 32 + lane) * 16], 0, 1);
    const v16h bf = *(const v16h*)&Wf[((size_t)(kt * 8 + wave) * 32 + lane) * 16];
#pragma unroll
    for (int mt = 0; mt < MPB; ++mt) {
      const v16h af = *(const v16h*)&Asm[kt & 1][mt * 512 + lane * 16];
      acc[mt] = __builtin_amdgcn_wmma_f32_16x16x32_f16(false, af, false, bf, (short)0, acc[mt], false, false);
    }
    __syncthreads();
  }
#else
  for (int kt = 0; kt < nK; ++kt) {
    if (kt + 1 < nK)
      __builtin_prefetch(&Wf[((size_t)((kt + 1) * 8 + wave) * 32 + lane) * 16], 0, 1);
    const v16h bf = *(const v16h*)&Wf[((size_t)(kt * 8 + wave) * 32 + lane) * 16];
#pragma unroll
    for (int mt = 0; mt < MPB; ++mt) {
      const v16h af = *(const v16h*)&Afr[((size_t)((mt0 + mt) * nK + kt) * 32 + lane) * 16];
      acc[mt] = __builtin_amdgcn_wmma_f32_16x16x32_f16(false, af, false, bf, (short)0, acc[mt], false, false);
    }
  }
#endif

  const int cc = wave * 16 + (lane & 15);
  const float bv = bias[cc];
  const int rsub = (lane >> 4) << 3;
  const int ktC = cc >> 5, k32 = cc & 31;
  const int lhC = (k32 >> 3) & 1, jC = (k32 & 7) + ((k32 >> 4) << 3);
#pragma unroll
  for (int mt = 0; mt < MPB; ++mt) {
    const int rbase = (mt0 + mt) * 16 + rsub;
#pragma unroll
    for (int i = 0; i < 8; ++i)
      C[(size_t)(rbase + i) * DIM + cc] = acc[mt][i] + bv;
    if (Cf) {  // also emit fragment-order f16 copy (output K-layout nK=4)
      const size_t fb = ((size_t)((mt0 + mt) * 4 + ktC) * 32 + (lhC << 4)) * 16 + jC;
#pragma unroll
      for (int i = 0; i < 8; ++i)
        Cf[fb + (size_t)(rsub + i) * 16] = (_Float16)(acc[mt][i] + bv);
    }
  }
}

// -------- fused MLP: Hout = X + (gelu(X@W1+b1))@W2 + b2 --------
// X read as fragment-order f16 from global; hidden written in phase-2 A-fragment
// order in LDS; epilogue emits f32 Hout plus fragment-order f16 copy for next layer.
__global__ void k_mlp_fused(const float* __restrict__ X, const _Float16* __restrict__ Xfr,
                            const _Float16* __restrict__ W1f, const float* __restrict__ b1,
                            const _Float16* __restrict__ W2f, const float* __restrict__ b2,
                            float* __restrict__ Hout, _Float16* __restrict__ Hf16) {
  __shared__ _Float16 Hf[16][32][16];   // 16x512 hidden, fragment order, 16 KB
  const int lane = threadIdx.x, wave = threadIdx.y;
  const int m0 = blockIdx.x * 16;
  const int rsub = (lane >> 4) << 3;

  // phase 1: hidden = gelu(X @ W1 + b1); 32 N-tiles -> 4 per wave
#pragma unroll
  for (int t = 0; t < 4; ++t) {
    const int nt = wave * 4 + t;
    v8f acc = {};
#pragma unroll
    for (int kt = 0; kt < 4; ++kt) {
      const v16h af = *(const v16h*)&Xfr[((size_t)(blockIdx.x * 4 + kt) * 32 + lane) * 16];
      const v16h bf = *(const v16h*)&W1f[((size_t)(kt * 32 + nt) * 32 + lane) * 16];
      acc = __builtin_amdgcn_wmma_f32_16x16x32_f16(false, af, false, bf, (short)0, acc, false, false);
    }
    const int cc = nt * 16 + (lane & 15);    // hidden column = phase-2 K index
    const float bv = b1[cc];
    const int kt2 = cc >> 5, k32 = cc & 31;
    const int l2 = afrag_lane(0, k32), j2 = afrag_j(k32);
#pragma unroll
    for (int i = 0; i < 8; ++i)
      Hf[kt2][l2 + rsub + i][j2] = (_Float16)gelu_f(acc[i] + bv);
  }
  __syncthreads();

  // phase 2: out = X + hidden @ W2 + b2 ; K = 512
  {
    v8f acc = {};
#pragma unroll
    for (int kt = 0; kt < 16; ++kt) {
      const v16h af = *(const v16h*)&Hf[kt][lane][0];
      const v16h bf = *(const v16h*)&W2f[((size_t)(kt * 8 + wave) * 32 + lane) * 16];
      acc = __builtin_amdgcn_wmma_f32_16x16x32_f16(false, af, false, bf, (short)0, acc, false, false);
    }
    const int cc = wave * 16 + (lane & 15);
    const float bv = b2[cc];
    const int ktC = cc >> 5, k32 = cc & 31;
    const int lhC = (k32 >> 3) & 1, jC = (k32 & 7) + ((k32 >> 4) << 3);
    const size_t fb = ((size_t)(blockIdx.x * 4 + ktC) * 32 + (lhC << 4)) * 16 + jC;
#pragma unroll
    for (int i = 0; i < 8; ++i) {
      const int gr = m0 + rsub + i;
      const float o = acc[i] + bv + X[(size_t)gr * DIM + cc];
      Hout[(size_t)gr * DIM + cc] = o;
      Hf16[fb + (size_t)(rsub + i) * 16] = (_Float16)o;
    }
  }
}

// ---------------- residual + LayerNorm: out = LN(xa + xb) (+ fragment f16 copy) ----------------
__global__ void k_add_ln(const float* __restrict__ xa, const float* __restrict__ xb,
                         const float* __restrict__ w, const float* __restrict__ b,
                         float* __restrict__ out, _Float16* __restrict__ outf) {
  const int node = blockIdx.x * 8 + threadIdx.y;
  const int lane = threadIdx.x;
  float v[4];
  float s = 0.0f;
#pragma unroll
  for (int i = 0; i < 4; ++i) {
    const int c = i * 32 + lane;
    v[i] = xa[(size_t)node * DIM + c] + xb[(size_t)node * DIM + c];
    s += v[i];
  }
  for (int off = 16; off > 0; off >>= 1) s += __shfl_xor(s, off, 32);
  const float mu = s * (1.0f / DIM);
  float q = 0.0f;
#pragma unroll
  for (int i = 0; i < 4; ++i) { const float d = v[i] - mu; q += d * d; }
  for (int off = 16; off > 0; off >>= 1) q += __shfl_xor(q, off, 32);
  const float rstd = rsqrtf(q * (1.0f / DIM) + 1e-5f);
  const int mt = node >> 4, row16 = node & 15;
  const int lf = afrag_lane(row16, lane & 31);   // k32 = lane
  const int jf = afrag_j(lane & 31);
#pragma unroll
  for (int i = 0; i < 4; ++i) {
    const int c = i * 32 + lane;
    const float o = (v[i] - mu) * rstd * w[c] + b[c];
    out[(size_t)node * DIM + c] = o;
    outf[((size_t)(mt * 4 + i) * 32 + lf) * 16 + jf] = (_Float16)o;  // kt = i
  }
}

// ---------------- edge attention: scores, segment softmax, aggregate ----------------

__global__ void k_scores(const float* __restrict__ q, const float* __restrict__ k,
                         const int* __restrict__ src, const int* __restrict__ dst,
                         float* __restrict__ sc, float* __restrict__ m) {
  int idx = blockIdx.x * blockDim.x + threadIdx.x;   // e*NH + h
  if (idx >= NE * NH) return;
  const int e = idx >> 2, hh = idx & 3;
  const int s = src[e], d = dst[e];
  const float* qp = q + (size_t)d * DIM + hh * HDM;
  const float* kp = k + (size_t)s * DIM + hh * HDM;
  float acc = 0.0f;
#pragma unroll
  for (int i = 0; i < HDM; i += 4) {
    float4 qa = *(const float4*)(qp + i);
    float4 kb = *(const float4*)(kp + i);
    acc += qa.x * kb.x + qa.y * kb.y + qa.z * kb.z + qa.w * kb.w;
  }
  acc *= 0.17677669529663687f;   // HD^-0.5
  sc[idx] = acc;
  atomicMaxF(&m[d * NH + hh], acc);
}

__global__ void k_expsum(const float* __restrict__ m, const int* __restrict__ dst,
                         float* __restrict__ sc, float* __restrict__ den) {
  int idx = blockIdx.x * blockDim.x + threadIdx.x;
  if (idx >= NE * NH) return;
  const int e = idx >> 2, hh = idx & 3;
  const int d = dst[e];
  const float ex = __expf(sc[idx] - m[d * NH + hh]);
  sc[idx] = ex;
  atomicAdd(&den[d * NH + hh], ex);
}

__global__ void k_agg(const float* __restrict__ ex, const float* __restrict__ den,
                      const float* __restrict__ v, const int* __restrict__ src,
                      const int* __restrict__ dst, float* __restrict__ agg) {
  int idx = blockIdx.x * blockDim.x + threadIdx.x;   // e*DIM + c
  if (idx >= NE * DIM) return;
  const int e = idx >> 7, c = idx & 127, hh = c >> 5;
  const int s = src[e], d = dst[e];
  const float w = ex[e * NH + hh] / den[d * NH + hh];
  atomicAdd(&agg[(size_t)d * DIM + c], w * v[(size_t)s * DIM + c]);
}

// ---------------- pooling + classifier head ----------------

__global__ void k_pool(const float* __restrict__ h, float* __restrict__ g, float* __restrict__ gout) {
  __shared__ float red[256];
  const int c = blockIdx.x, t = threadIdx.x;
  float s = 0.0f;
  for (int r = t; r < NN; r += 256) s += h[(size_t)r * DIM + c];
  red[t] = s; __syncthreads();
  for (int off = 128; off > 0; off >>= 1) { if (t < off) red[t] += red[t + off]; __syncthreads(); }
  if (t == 0) { const float gm = red[0] * (1.0f / NN); g[c] = gm; gout[c] = gm; }
}

__global__ void k_classifier(const float* __restrict__ g,
                             const float* __restrict__ lnw, const float* __restrict__ lnb,
                             const float* __restrict__ cw1, const float* __restrict__ cb1,
                             const float* __restrict__ cw2, const float* __restrict__ cb2,
                             const float* __restrict__ cw,  const float* __restrict__ cb,
                             float* __restrict__ logits) {
  __shared__ float z[DIM], z2[2 * DIM], z3[DIM], red[2];
  const int t = threadIdx.x;
  if (t == 0) {
    float mu = 0.0f;
    for (int i = 0; i < DIM; ++i) mu += g[i];
    mu *= (1.0f / DIM);
    float va = 0.0f;
    for (int i = 0; i < DIM; ++i) { const float d = g[i] - mu; va += d * d; }
    red[0] = mu; red[1] = rsqrtf(va * (1.0f / DIM) + 1e-5f);
  }
  __syncthreads();
  if (t < DIM) z[t] = (g[t] - red[0]) * red[1] * lnw[t] + lnb[t];
  __syncthreads();
  { float a = cb1[t];
#pragma unroll 4
    for (int k = 0; k < DIM; ++k) a += z[k] * cw1[k * (2 * DIM) + t];
    z2[t] = gelu_f(a); }
  __syncthreads();
  if (t < DIM) { float a = cb2[t];
#pragma unroll 4
    for (int k = 0; k < 2 * DIM; ++k) a += z2[k] * cw2[k * DIM + t];
    z3[t] = a; }
  __syncthreads();
  if (t < NCLS) { float a = cb[t];
#pragma unroll 4
    for (int k = 0; k < DIM; ++k) a += z3[k] * cw[k * NCLS + t];
    logits[t] = a; }
}

// ---------------- host orchestration ----------------

extern "C" void kernel_launch(void* const* d_in, const int* in_sizes, int n_in,
                              void* d_out, int out_size, void* d_ws, size_t ws_size,
                              hipStream_t stream) {
  (void)in_sizes; (void)n_in; (void)out_size; (void)ws_size;
  const float* node_features = (const float*)d_in[0];
  const int*   edge_index    = (const int*)d_in[1];
  const float* node_emb_w = (const float*)d_in[3];
  const float* node_emb_b = (const float*)d_in[4];
  const float* qw  = (const float*)d_in[7];
  const float* qb  = (const float*)d_in[8];
  const float* kw  = (const float*)d_in[9];
  const float* kbv = (const float*)d_in[10];
  const float* vw  = (const float*)d_in[11];
  const float* vb  = (const float*)d_in[12];
  const float* ow  = (const float*)d_in[13];
  const float* ob  = (const float*)d_in[14];
  const float* lnw = (const float*)d_in[15];
  const float* lnb = (const float*)d_in[16];
  const float* mw1 = (const float*)d_in[17];
  const float* mb1 = (const float*)d_in[18];
  const float* mw2 = (const float*)d_in[19];
  const float* mb2 = (const float*)d_in[20];
  const float* cls_ln_w = (const float*)d_in[21];
  const float* cls_ln_b = (const float*)d_in[22];
  const float* cw1 = (const float*)d_in[23];
  const float* cb1 = (const float*)d_in[24];
  const float* cw2 = (const float*)d_in[25];
  const float* cb2 = (const float*)d_in[26];
  const float* cls_w = (const float*)d_in[27];
  const float* cls_b = (const float*)d_in[28];

  const int* srcI = edge_index;
  const int* dstI = edge_index + NE;

  char* ws = (char*)d_ws;
  size_t off = 0;
  auto carveF = [&](size_t n) { float* p = (float*)(ws + off); off += n * sizeof(float); return p; };
  float* h    = carveF((size_t)NN * DIM);
  float* bufQ = carveF((size_t)NN * DIM);   // q, then agg
  float* bufK = carveF((size_t)NN * DIM);   // k, then out-proj result
  float* bufV = carveF((size_t)NN * DIM);   // v, then out_ln
  float* eh   = carveF((size_t)NE * NH);
  float* mA   = carveF((size_t)NN * NH);
  float* dA   = carveF((size_t)NN * NH);
  float* g    = carveF(DIM);
  auto carveH = [&](size_t n) { _Float16* p = (_Float16*)(ws + off); off += n * sizeof(_Float16); return p; };
  _Float16* hf    = carveH((size_t)NN * DIM);   // h, fragment order
  _Float16* xf    = carveH((size_t)NN * DIM);   // out_ln, fragment order
  _Float16* aggf  = carveH((size_t)NN * DIM);   // agg, fragment order
  _Float16* nff   = carveH((size_t)NN * NFI);   // node_features, fragment order
  _Float16* emb16 = carveH((size_t)NFI * DIM);
  _Float16* qw16  = carveH((size_t)NL * DIM * DIM);
  _Float16* kw16  = carveH((size_t)NL * DIM * DIM);
  _Float16* vw16  = carveH((size_t)NL * DIM * DIM);
  _Float16* ow16  = carveH((size_t)NL * DIM * DIM);
  _Float16* m116  = carveH((size_t)NL * DIM * 4 * DIM);
  _Float16* m216  = carveH((size_t)NL * 4 * DIM * DIM);

  const dim3 wblk(32, 8);
  auto swz = [&](const float* s, _Float16* d, int K, int N) {
    k_swizzle_w<<<(K * N + 255) / 256, 256, 0, stream>>>(s, d, K, N);
  };
  swz(node_emb_w, emb16, NFI, DIM);
  for (int i = 0; i < NL; ++i) {
    swz(qw  + (size_t)i * DIM * DIM,     qw16 + (size_t)i * DIM * DIM,     DIM, DIM);
    swz(kw  + (size_t)i * DIM * DIM,     kw16 + (size_t)i * DIM * DIM,     DIM, DIM);
    swz(vw  + (size_t)i * DIM * DIM,     vw16 + (size_t)i * DIM * DIM,     DIM, DIM);
    swz(ow  + (size_t)i * DIM * DIM,     ow16 + (size_t)i * DIM * DIM,     DIM, DIM);
    swz(mw1 + (size_t)i * DIM * 4 * DIM, m116 + (size_t)i * DIM * 4 * DIM, DIM, 4 * DIM);
    swz(mw2 + (size_t)i * 4 * DIM * DIM, m216 + (size_t)i * 4 * DIM * DIM, 4 * DIM, DIM);
  }

  // GEMM launcher: 312 blocks of 4 M-tiles + 1 tail block of 2 (1250 tiles total)
  auto gemm = [&](const _Float16* Afr, const _Float16* Wf, const float* bias,
                  float* C, _Float16* Cf, int K) {
    k_gemm_wmma<4><<<312, wblk, 0, stream>>>(Afr, Wf, bias, C, Cf, K, 0);
    k_gemm_wmma<2><<<1,   wblk, 0, stream>>>(Afr, Wf, bias, C, Cf, K, 1248);
  };

  // node embedding: h = nf @ emb_w + b (also produce hf fragment copy)
  k_cvt_frag<<<(NN * NFI + 255) / 256, 256, 0, stream>>>(node_features, nff, NFI, NN * NFI);
  gemm(nff, emb16, node_emb_b, h, hf, NFI);

  for (int i = 0; i < NL; ++i) {
    gemm(hf, qw16 + (size_t)i * DIM * DIM, qb  + i * DIM, bufQ, nullptr, DIM);
    gemm(hf, kw16 + (size_t)i * DIM * DIM, kbv + i * DIM, bufK, nullptr, DIM);
    gemm(hf, vw16 + (size_t)i * DIM * DIM, vb  + i * DIM, bufV, nullptr, DIM);
    k_fill<<<(NN * NH + 255) / 256, 256, 0, stream>>>(mA, -INFINITY, NN * NH);
    k_fill<<<(NN * NH + 255) / 256, 256, 0, stream>>>(dA, 0.0f, NN * NH);
    k_scores<<<(NE * NH + 255) / 256, 256, 0, stream>>>(bufQ, bufK, srcI, dstI, eh, mA);
    k_expsum<<<(NE * NH + 255) / 256, 256, 0, stream>>>(mA, dstI, eh, dA);
    k_fill<<<(NN * DIM + 255) / 256, 256, 0, stream>>>(bufQ, 0.0f, NN * DIM);      // bufQ -> agg
    k_agg<<<(NE * DIM + 255) / 256, 256, 0, stream>>>(eh, dA, bufV, srcI, dstI, bufQ);
    k_cvt_frag<<<(NN * DIM + 255) / 256, 256, 0, stream>>>(bufQ, aggf, DIM, NN * DIM);
    gemm(aggf, ow16 + (size_t)i * DIM * DIM, ob + i * DIM, bufK, nullptr, DIM);
    k_add_ln<<<NN / 8, wblk, 0, stream>>>(bufK, h, lnw + i * DIM, lnb + i * DIM, bufV, xf);
    k_mlp_fused<<<NN / 16, wblk, 0, stream>>>(bufV, xf,
                                              m116 + (size_t)i * DIM * 4 * DIM, mb1 + i * 4 * DIM,
                                              m216 + (size_t)i * 4 * DIM * DIM, mb2 + i * DIM,
                                              h, hf);
  }

  float* outF = (float*)d_out;
  k_pool<<<DIM, 256, 0, stream>>>(h, g, outF + 10 + (size_t)NN * DIM);
  k_copy<<<(NN * DIM + 255) / 256, 256, 0, stream>>>(h, outF + 10, NN * DIM);
  k_classifier<<<1, 256, 0, stream>>>(g, cls_ln_w, cls_ln_b, cw1, cb1, cw2, cb2, cls_w, cls_b, outF);
}
